// VectorQuantizer_62423054680143
// MI455X (gfx1250) — matrix operand order repaired and checked by reading, not compiled, not run on hardware
//
#include <hip/hip_runtime.h>

#define NROWS 131072
#define DIM   128
#define KEMB  1024

#define CHUNK     64                      // embeddings per LDS chunk
#define NCHUNK    (KEMB / CHUNK)          // 16
#define ROWBYTES  272                     // 256B row + 16B TDM pad (bank-conflict-free)
#define BUFBYTES  (CHUNK * ROWBYTES)      // 17408 per buffer (hi or lo)

typedef __attribute__((ext_vector_type(16))) _Float16 v16h;
typedef __attribute__((ext_vector_type(8)))  float    v8f;
typedef __attribute__((ext_vector_type(4)))  unsigned v4u;
typedef __attribute__((ext_vector_type(8)))  int      v8i;
typedef __attribute__((ext_vector_type(4)))  int      v4i;

union HFrag {
    v16h     v;
    _Float16 e[16];
    uint4    q[2];
};

// ---------------------------------------------------------------------------
// TDM: load a 64x128 f16 tile (row stride 128 elems) global -> LDS, padding
// 4 DWORDs after every 64 DWORDs (so LDS row stride = 272B).
// D# group0: count=1, lds_addr, global_addr, type=2.
// D# group1: data_size=1(2B), pad_enable, pad_interval=64dw(code5),
//            pad_amount=4dw(code3); tensor_dim0=128, tensor_dim1=1024,
//            tile_dim0=128, tile_dim1=64, tensor_dim0_stride=128.
// ---------------------------------------------------------------------------
__device__ __forceinline__ void tdm_load_64x128_f16(const _Float16* src,
                                                    unsigned lds_byte_addr)
{
    unsigned long long ga = (unsigned long long)(uintptr_t)src;
    v4u g0 = { 1u,                                   // count=1
               lds_byte_addr,                        // LDS dest
               (unsigned)ga,                         // global addr [31:0]
               ((unsigned)(ga >> 32) & 0x01ffffffu) | 0x80000000u }; // [56:32] | type=2
    v8i g1 = { (int)0x07510000,   // data_size=1, pad_en, interval=64dw, amount=4dw
               (int)0x00800000,   // tensor_dim0 = 128 (bits 63:48)
               (int)0x04000000,   // tensor_dim1 = 1024 (bits 95:80)
               (int)0x00800000,   // tile_dim0 = 128 (bits 127:112)
               64,                // tile_dim1 = 64
               128,               // tensor_dim0_stride = 128
               0, 0 };
    v4i z4 = {0, 0, 0, 0};
#if defined(__clang_major__) && (__clang_major__ >= 23)
    v8i z8 = {0, 0, 0, 0, 0, 0, 0, 0};
    __builtin_amdgcn_tensor_load_to_lds(g0, g1, z4, z4, z8, 0);
#else
    __builtin_amdgcn_tensor_load_to_lds(g0, g1, z4, z4, 0);
#endif
}

// ---------------------------------------------------------------------------
// Kernel 1: split embeddings into f16 hi/lo, compute ||e_k||^2.
// ---------------------------------------------------------------------------
__global__ __launch_bounds__(32)
void vq_prep(const float* __restrict__ emb,
             _Float16* __restrict__ ehi, _Float16* __restrict__ elo,
             float* __restrict__ enorm)
{
    const int k    = blockIdx.x;
    const int lane = threadIdx.x;
    const float4 p = *(const float4*)(emb + (size_t)k * DIM + 4 * lane);
    float vs[4] = {p.x, p.y, p.z, p.w};
    _Float16* dh = ehi + (size_t)k * DIM + 4 * lane;
    _Float16* dl = elo + (size_t)k * DIM + 4 * lane;
    float ss = 0.0f;
#pragma unroll
    for (int i = 0; i < 4; ++i) {
        ss += vs[i] * vs[i];
        _Float16 h = (_Float16)vs[i];
        dh[i] = h;
        dl[i] = (_Float16)(vs[i] - (float)h);
    }
#pragma unroll
    for (int off = 16; off >= 1; off >>= 1) ss += __shfl_xor(ss, off, 32);
    if (lane == 0) enorm[k] = ss;
}

// ---------------------------------------------------------------------------
// Kernel 2: main VQ kernel. 256 threads = 8 waves; each wave owns 16 rows.
// Embedding chunks staged to LDS by the Tensor Data Mover, double-buffered.
// Distances via split-precision f16 WMMA: x.e ~= xh.eh + xl.eh + xh.el.
// ---------------------------------------------------------------------------
__global__ __launch_bounds__(256)
void vq_main(const float* __restrict__ x, const float* __restrict__ emb,
             const _Float16* __restrict__ ehi, const _Float16* __restrict__ elo,
             const float* __restrict__ enorm,
             float* __restrict__ out, float* __restrict__ partials)
{
    __shared__ char  smem[4 * BUFBYTES];   // {buf0,buf1} x {hi,lo}
    __shared__ int   sIdx[8][16];
    __shared__ float sPart[8];

    const int tid    = threadIdx.x;
    const int wave   = tid >> 5;
    const int lane   = tid & 31;
    const int lane16 = lane & 15;
    const int half   = lane >> 4;

    const int rowBase = blockIdx.x * 128 + wave * 16;
    const int myRow   = rowBase + lane16;

    // Prologue: TDM-prefetch chunks 0 and 1 (hi+lo each) into both buffers.
    if (wave == 0) {
        const unsigned b0 = (unsigned)(uintptr_t)&smem[0];
        tdm_load_64x128_f16(ehi,               b0);
        tdm_load_64x128_f16(elo,               b0 + BUFBYTES);
        tdm_load_64x128_f16(ehi + CHUNK * DIM, b0 + 2 * BUFBYTES);
        tdm_load_64x128_f16(elo + CHUNK * DIM, b0 + 3 * BUFBYTES);
    }

    // ---- Build A fragments (16x32 f16, hi & lo) for 4 depth chunks -------
    // A layout (ISA 7.12.2, 16-bit A 16x32): lane%16 = row M, half selects
    // depth runs {8h..8h+7} U {16+8h..16+8h+7} within each 32-chunk.
    HFrag Ahi[4], Alo[4];
    const float* xr = x + (size_t)myRow * DIM;
#pragma unroll
    for (int c = 0; c < 4; ++c) {
        const int d0 = 32 * c + 8 * half;
        float t[16];
        float4 p;
        p = *(const float4*)(xr + d0);      t[0]=p.x;  t[1]=p.y;  t[2]=p.z;  t[3]=p.w;
        p = *(const float4*)(xr + d0 + 4);  t[4]=p.x;  t[5]=p.y;  t[6]=p.z;  t[7]=p.w;
        p = *(const float4*)(xr + d0 + 16); t[8]=p.x;  t[9]=p.y;  t[10]=p.z; t[11]=p.w;
        p = *(const float4*)(xr + d0 + 20); t[12]=p.x; t[13]=p.y; t[14]=p.z; t[15]=p.w;
#pragma unroll
        for (int i = 0; i < 16; ++i) {
            _Float16 h = (_Float16)t[i];
            Ahi[c].e[i] = h;
            Alo[c].e[i] = (_Float16)(t[i] - (float)h);
        }
    }

    float bestV[8];
    int   bestI[8];
#pragma unroll
    for (int r = 0; r < 8; ++r) { bestV[r] = 3.402823466e38f; bestI[r] = 0; }

    // ---- Chunk loop: compute on buf[kc&1] while TDM fills the other ------
    for (int kc = 0; kc < NCHUNK; ++kc) {
        if (wave == 0) {
            if (kc + 1 < NCHUNK) __builtin_amdgcn_s_wait_tensorcnt(2);
            else                 __builtin_amdgcn_s_wait_tensorcnt(0);
        }
        __syncthreads();   // chunk kc resident in buf[kc&1] for all waves

        const char* hiBuf = smem + (kc & 1) * 2 * BUFBYTES;
        const char* loBuf = hiBuf + BUFBYTES;

#pragma unroll
        for (int pr = 0; pr < 2; ++pr) {          // two 16-col tile pairs / chunk
            const int r0  = 32 * pr + lane16;     // local embedding row (tile A)
            const int r1  = r0 + 16;              // local embedding row (tile B)
            const int n0  = kc * CHUNK + r0;      // global k (C/D: N = lane%16)
            const int n1  = n0 + 16;
            v8f acc0 = {0.f,0.f,0.f,0.f,0.f,0.f,0.f,0.f};
            v8f acc1 = {0.f,0.f,0.f,0.f,0.f,0.f,0.f,0.f};
#pragma unroll
            for (int c = 0; c < 4; ++c) {
                // B layout: lane n holds column n, depth runs
                // {8h..8h+7} U {16+8h..16+8h+7} of the 32-chunk.
                const int db2 = (32 * c + 8 * half) * 2;    // byte offset in row
                HFrag b0h, b0l, b1h, b1l;
                const uint4* p0h = (const uint4*)(hiBuf + r0 * ROWBYTES + db2);
                const uint4* p1h = (const uint4*)(hiBuf + r1 * ROWBYTES + db2);
                const uint4* p0l = (const uint4*)(loBuf + r0 * ROWBYTES + db2);
                const uint4* p1l = (const uint4*)(loBuf + r1 * ROWBYTES + db2);
                b0h.q[0] = p0h[0]; b0h.q[1] = p0h[2];
                b1h.q[0] = p1h[0]; b1h.q[1] = p1h[2];
                b0l.q[0] = p0l[0]; b0l.q[1] = p0l[2];
                b1l.q[0] = p1l[0]; b1l.q[1] = p1l[2];

                acc0 = __builtin_amdgcn_wmma_f32_16x16x32_f16(false, Ahi[c].v, false, b0h.v, (short)0, acc0, false, false);
                acc1 = __builtin_amdgcn_wmma_f32_16x16x32_f16(false, Ahi[c].v, false, b1h.v, (short)0, acc1, false, false);
                acc0 = __builtin_amdgcn_wmma_f32_16x16x32_f16(false, Alo[c].v, false, b0h.v, (short)0, acc0, false, false);
                acc1 = __builtin_amdgcn_wmma_f32_16x16x32_f16(false, Alo[c].v, false, b1h.v, (short)0, acc1, false, false);
                acc0 = __builtin_amdgcn_wmma_f32_16x16x32_f16(false, Ahi[c].v, false, b0l.v, (short)0, acc0, false, false);
                acc1 = __builtin_amdgcn_wmma_f32_16x16x32_f16(false, Ahi[c].v, false, b1l.v, (short)0, acc1, false, false);
            }
            const float en0 = enorm[n0];
            const float en1 = enorm[n1];
#pragma unroll
            for (int r = 0; r < 8; ++r) {         // row m = 8*half + r
                const float s0 = en0 - 2.0f * acc0[r];
                if (s0 < bestV[r]) { bestV[r] = s0; bestI[r] = n0; }
                const float s1 = en1 - 2.0f * acc1[r];
                if (s1 < bestV[r]) { bestV[r] = s1; bestI[r] = n1; }
            }
        }

        __syncthreads();   // all waves done reading buf[kc&1]
        if (wave == 0 && kc + 2 < NCHUNK) {
            const unsigned bo = (unsigned)(uintptr_t)&smem[(kc & 1) * 2 * BUFBYTES];
            tdm_load_64x128_f16(ehi + (size_t)(kc + 2) * CHUNK * DIM, bo);
            tdm_load_64x128_f16(elo + (size_t)(kc + 2) * CHUNK * DIM, bo + BUFBYTES);
        }
    }

    // ---- Cross-lane argmin reduce within each 16-lane group ---------------
#pragma unroll
    for (int r = 0; r < 8; ++r) {
        float v = bestV[r];
        int   i = bestI[r];
#pragma unroll
        for (int m = 8; m >= 1; m >>= 1) {   // masks < 16: stays in group
            const float ov = __shfl_xor(v, m, 32);
            const int   oi = __shfl_xor(i, m, 32);
            if (ov < v || (ov == v && oi < i)) { v = ov; i = oi; }
        }
        bestI[r] = i;
    }
    if (lane16 == 0) {
#pragma unroll
        for (int r = 0; r < 8; ++r) sIdx[wave][8 * half + r] = bestI[r];
    }
    __syncthreads();

    // ---- Gather, straight-through write, loss partial ---------------------
    float lsum = 0.0f;
    const int d4 = 4 * lane;
    for (int r = 0; r < 16; ++r) {
        const int row = rowBase + r;
        const int idx = sIdx[wave][r];
        const float4 q  = *(const float4*)(emb + (size_t)idx * DIM + d4);
        const float4 xv = *(const float4*)(x   + (size_t)row * DIM + d4);
        const float dx = q.x - xv.x, dy = q.y - xv.y, dz = q.z - xv.z, dw = q.w - xv.w;
        float4 o;  // x + (q - x): matches straight-through rounding
        o.x = xv.x + dx; o.y = xv.y + dy; o.z = xv.z + dz; o.w = xv.w + dw;
        *(float4*)(out + (size_t)row * DIM + d4) = o;
        lsum += dx * dx + dy * dy + dz * dz + dw * dw;
    }
#pragma unroll
    for (int m = 16; m >= 1; m >>= 1) lsum += __shfl_xor(lsum, m, 32);
    if (lane == 0) sPart[wave] = lsum;
    __syncthreads();
    if (tid == 0) {
        float t = 0.0f;
#pragma unroll
        for (int w = 0; w < 8; ++w) t += sPart[w];
        partials[blockIdx.x] = t;   // plain store: deterministic
    }
}

// ---------------------------------------------------------------------------
// Kernel 3: fixed-order sum of block partials -> vq_loss = 2 * mean
// ---------------------------------------------------------------------------
__global__ void vq_finish(const float* __restrict__ partials, float* __restrict__ out)
{
    if (threadIdx.x == 0 && blockIdx.x == 0) {
        float s = 0.0f;
        for (int i = 0; i < NROWS / 128; ++i) s += partials[i];
        const float mse = s / (float)((size_t)NROWS * DIM);
        out[(size_t)NROWS * DIM] = 2.0f * mse;
    }
}

extern "C" void kernel_launch(void* const* d_in, const int* in_sizes, int n_in,
                              void* d_out, int out_size, void* d_ws, size_t ws_size,
                              hipStream_t stream)
{
    const float* x   = (const float*)d_in[0];   // [131072,128] f32
    const float* emb = (const float*)d_in[1];   // [1024,128]   f32

    _Float16* ehi   = (_Float16*)d_ws;                    // 1024*128 f16
    _Float16* elo   = ehi + (size_t)KEMB * DIM;           // 1024*128 f16
    float*    enorm = (float*)(elo + (size_t)KEMB * DIM); // 1024 f32
    float*    parts = enorm + KEMB;                       // 1024 f32

    float* out = (float*)d_out;   // [131072*128] quantized_st, then [1] vq_loss

    vq_prep  <<<KEMB,        32, 0, stream>>>(emb, ehi, elo, enorm);
    vq_main  <<<NROWS / 128, 256, 0, stream>>>(x, emb, ehi, elo, enorm, out, parts);
    vq_finish<<<1,           32, 0, stream>>>(parts, out);
}
